// Swinv2_40183714021975
// MI455X (gfx1250) — matrix operand order repaired and checked by reading, not compiled, not run on hardware
//
#include <hip/hip_runtime.h>

#define DIM   384
#define HEADS 12
#define HD    32
#define WS    24
#define NTOK  576            // tokens per window
#define HH    48
#define WWID  48
#define BATCH 4
#define TW    16             // total windows = B * 4
#define MLPD  1536
#define HID   512
#define TAB   2209           // (2*WS-1)^2
#define TOKS  9216           // BATCH*HH*WWID
#define LOG100 4.6051701859880914f

typedef __bf16 bf16_t;
typedef bf16_t v16bf __attribute__((ext_vector_type(16)));
typedef float  v8f   __attribute__((ext_vector_type(8)));

__device__ inline unsigned short f2bf(float f) {
    unsigned int u = __float_as_uint(f);
    unsigned int r = (u + 0x7fffu + ((u >> 16) & 1u)) >> 16;   // RNE
    return (unsigned short)r;
}

union FragU { uint4 u[2]; v16bf v; };
union Pack8 { unsigned short s[8]; uint4 u; };

// A-matrix fragment (16x32 bf16): lane holds row lane%16, 16 contiguous K
__device__ inline v16bf load_fragA(const unsigned short* p) {
    FragU f;
    f.u[0] = *reinterpret_cast<const uint4*>(p);
    f.u[1] = *reinterpret_cast<const uint4*>(p + 8);
    return f.v;
}
// B-matrix fragment (32x16 bf16, stored [N][K]): lane holds col lane%16, 16 contiguous K
__device__ inline v16bf load_fragB(const unsigned short* p) {
    FragU f;
    f.u[0] = *reinterpret_cast<const uint4*>(p);
    f.u[1] = *reinterpret_cast<const uint4*>(p + 8);
    return f.v;
}

__device__ inline v8f wmma_bf16(v16bf a, v16bf b, v8f c) {
    return __builtin_amdgcn_wmma_f32_16x16x32_bf16(false, a, false, b, (short)0, c, false, false);
}

// ---------------- elementwise / setup kernels ----------------

__global__ void k_cvt_bf16(const float* __restrict__ s, unsigned short* __restrict__ d, int n) {
    int i = blockIdx.x * blockDim.x + threadIdx.x;
    if (i < n) d[i] = f2bf(s[i]);
}

// continuous position bias MLP: 2 -> 512 -> 12, then 16*sigmoid
__global__ void k_cpb(const float* __restrict__ w1, const float* __restrict__ b1,
                      const float* __restrict__ w2, float* __restrict__ btab) {
    int t = blockIdx.x * blockDim.x + threadIdx.x;
    if (t >= TAB) return;
    float r0 = (float)(t / (2 * WS - 1) - (WS - 1));
    float r1 = (float)(t % (2 * WS - 1) - (WS - 1));
    const float sc  = 8.0f / 11.0f;        // /(PWS-1)*8 with PWS=12
    const float il8 = 1.0f / 3.0f;         // 1/log2(8)
    float x0 = r0 * sc, x1 = r1 * sc;
    float c0 = copysignf(log2f(fabsf(x0) + 1.f) * il8, x0);
    float c1 = copysignf(log2f(fabsf(x1) + 1.f) * il8, x1);
    float acc[HEADS];
    for (int i = 0; i < HEADS; ++i) acc[i] = 0.f;
    for (int j = 0; j < HID; ++j) {
        float hv = fmaxf(w1[j * 2] * c0 + w1[j * 2 + 1] * c1 + b1[j], 0.f);
        for (int i = 0; i < HEADS; ++i) acc[i] += hv * w2[i * HID + j];
    }
    for (int i = 0; i < HEADS; ++i)
        btab[t * HEADS + i] = 16.f / (1.f + __expf(-acc[i]));
}

// expand bias table through REL_IDX into [12][576][576]
__global__ void k_bias_expand(const float* __restrict__ btab, float* __restrict__ biasF) {
    int idx = blockIdx.x * blockDim.x + threadIdx.x;
    if (idx >= HEADS * NTOK * NTOK) return;
    int h = idx / (NTOK * NTOK);
    int r = idx % (NTOK * NTOK);
    int n = r / NTOK, m = r % NTOK;
    int qi = n / WS, qj = n % WS, ki = m / WS, kj = m % WS;
    int rel = (qi - ki + WS - 1) * (2 * WS - 1) + (qj - kj + WS - 1);
    biasF[idx] = btab[rel * HEADS + h];
}

// roll + window partition + f32->bf16
__global__ void k_stage_win(const float* __restrict__ x, unsigned short* __restrict__ o, int s) {
    int idx = blockIdx.x * blockDim.x + threadIdx.x;
    if (idx >= TW * NTOK * DIM) return;
    int c = idx % DIM; int t = idx / DIM;
    int win = t / NTOK, n = t % NTOK;
    int b = win >> 2, wrow = (win >> 1) & 1, wcol = win & 1;
    int gh = (wrow * WS + n / WS + s) % HH;
    int gw = (wcol * WS + n % WS + s) % WWID;
    o[idx] = f2bf(x[((size_t)((b * HH + gh) * WWID + gw)) * DIM + c]);
}

// ---------------- WMMA GEMM kernels ----------------

// QKV projection: xw[9216][384] @ qkv_w[1152][384]^T, 32x32 tile per wave,
// scatter to q/k raw (f32) and V^T (bf16)
__global__ void k_qkv(const unsigned short* __restrict__ A, const unsigned short* __restrict__ Bw,
                      const float* __restrict__ bias, float* __restrict__ qraw,
                      float* __restrict__ kraw, unsigned short* __restrict__ vT) {
    int nt = blockIdx.x;                 // 0..35 : output chunk of 32 (q heads, k heads, v heads)
    int bm = blockIdx.y;                 // 0..287 : tile of 32 tokens
    int lane = threadIdx.x, half = lane >> 4, lc = lane & 15;
    const unsigned short* arow0 = A + ((size_t)(bm * 32 + lc)) * DIM + half * 16;
    const unsigned short* arow1 = arow0 + (size_t)16 * DIM;
    const unsigned short* brow0 = Bw + ((size_t)(nt * 32 + lc)) * DIM + half * 16;
    const unsigned short* brow1 = brow0 + (size_t)16 * DIM;
    v8f acc[2][2];
    #pragma unroll
    for (int mi = 0; mi < 2; ++mi)
        #pragma unroll
        for (int j = 0; j < 2; ++j)
            #pragma unroll
            for (int r = 0; r < 8; ++r) acc[mi][j][r] = 0.f;
    for (int k0 = 0; k0 < DIM; k0 += 32) {
        v16bf a0 = load_fragA(arow0 + k0);
        v16bf a1 = load_fragA(arow1 + k0);
        v16bf b0 = load_fragB(brow0 + k0);
        v16bf b1 = load_fragB(brow1 + k0);
        acc[0][0] = wmma_bf16(a0, b0, acc[0][0]);
        acc[0][1] = wmma_bf16(a0, b1, acc[0][1]);
        acc[1][0] = wmma_bf16(a1, b0, acc[1][0]);
        acc[1][1] = wmma_bf16(a1, b1, acc[1][1]);
    }
    int w = bm / 18;                    // window (576/32 = 18 tiles per window)
    int tok0 = (bm % 18) * 32;
    int kind = nt / 12, h = nt % 12;
    float bi0 = bias[nt * 32 + lc], bi1 = bias[nt * 32 + 16 + lc];
    if (kind < 2) {
        float* dst = (kind == 0) ? qraw : kraw;
        size_t base = ((size_t)(w * HEADS + h)) * NTOK * HD;
        #pragma unroll
        for (int mi = 0; mi < 2; ++mi)
            #pragma unroll
            for (int r = 0; r < 8; ++r) {
                int tok = tok0 + mi * 16 + r + 8 * half;
                dst[base + (size_t)tok * HD + lc]      = acc[mi][0][r] + bi0;
                dst[base + (size_t)tok * HD + 16 + lc] = acc[mi][1][r] + bi1;
            }
    } else {  // V stored transposed: [w][h][d][tok]
        size_t base = ((size_t)(w * HEADS + h)) * HD * NTOK;
        #pragma unroll
        for (int mi = 0; mi < 2; ++mi) {
            Pack8 p0, p1;
            #pragma unroll
            for (int r = 0; r < 8; ++r) {
                p0.s[r] = f2bf(acc[mi][0][r] + bi0);
                p1.s[r] = f2bf(acc[mi][1][r] + bi1);
            }
            int t0 = tok0 + mi * 16 + 8 * half;   // 8 consecutive tokens -> one b128 store
            *reinterpret_cast<uint4*>(vT + base + (size_t)lc        * NTOK + t0) = p0.u;
            *reinterpret_cast<uint4*>(vT + base + (size_t)(16 + lc) * NTOK + t0) = p1.u;
        }
    }
}

// L2-normalize q/k rows (d=32); fold exp(min(logit_scale,ln100)) into q
__global__ void k_norm_qk(const float* __restrict__ src, unsigned short* __restrict__ dst,
                          const float* __restrict__ lsc) {
    int row = blockIdx.x * blockDim.x + threadIdx.x;
    if (row >= TW * HEADS * NTOK) return;
    const float* p = src + (size_t)row * HD;
    float ss = 0.f;
    for (int i = 0; i < HD; ++i) { float x = p[i]; ss += x * x; }
    float f = 1.f / fmaxf(sqrtf(ss), 1e-12f);
    if (lsc) { int h = (row / NTOK) % HEADS; f *= __expf(fminf(lsc[h], LOG100)); }
    unsigned short* q = dst + (size_t)row * HD;
    for (int i = 0; i < HD; ++i) q[i] = f2bf(p[i] * f);
}

// flash-style attention: one wave = (window, head, 16-query tile)
__global__ void k_attn(const unsigned short* __restrict__ qn, const unsigned short* __restrict__ kn,
                       const unsigned short* __restrict__ vT, const float* __restrict__ biasF,
                       unsigned short* __restrict__ obuf, int shift) {
    __shared__ __align__(16) unsigned short Pl[16 * 32];
    int qt = blockIdx.x, h = blockIdx.y, w = blockIdx.z;
    int lane = threadIdx.x, half = lane >> 4, lc = lane & 15;
    size_t qkbase = ((size_t)(w * HEADS + h)) * NTOK * HD;
    v16bf qA = load_fragA(qn + qkbase + (size_t)(qt * 16 + lc) * HD + half * 16);
    float mx[8], li[8]; v8f o0, o1;
    for (int r = 0; r < 8; ++r) { mx[r] = -1e30f; li[r] = 0.f; o0[r] = 0.f; o1[r] = 0.f; }
    int wrow = (w >> 1) & 1, wcol = w & 1;
    int qrid[8];
    if (shift) {
        for (int r = 0; r < 8; ++r) {
            int tok = qt * 16 + r + 8 * half;
            int gh = wrow * WS + tok / WS, gw = wcol * WS + tok % WS;
            qrid[r] = (gh < 24 ? 0 : (gh < 36 ? 1 : 2)) * 3 + (gw < 24 ? 0 : (gw < 36 ? 1 : 2));
        }
    }
    size_t vbase = ((size_t)(w * HEADS + h)) * HD * NTOK;
    size_t bbase = (size_t)h * NTOK * NTOK + (size_t)(qt * 16) * NTOK;
    for (int kb = 0; kb < NTOK / 32; ++kb) {
        v16bf kB0 = load_fragB(kn + qkbase + (size_t)(kb * 32 + lc) * HD + half * 16);
        v16bf kB1 = load_fragB(kn + qkbase + (size_t)(kb * 32 + 16 + lc) * HD + half * 16);
        v8f z; for (int r = 0; r < 8; ++r) z[r] = 0.f;
        v8f s0 = wmma_bf16(qA, kB0, z);
        v8f s1 = wmma_bf16(qA, kB1, z);
        int kr0 = 0, kr1 = 0;
        if (shift) {
            int kt0 = kb * 32 + lc, kt1 = kt0 + 16;
            int gh0 = wrow * WS + kt0 / WS, gw0 = wcol * WS + kt0 % WS;
            int gh1 = wrow * WS + kt1 / WS, gw1 = wcol * WS + kt1 % WS;
            kr0 = (gh0 < 24 ? 0 : (gh0 < 36 ? 1 : 2)) * 3 + (gw0 < 24 ? 0 : (gw0 < 36 ? 1 : 2));
            kr1 = (gh1 < 24 ? 0 : (gh1 < 36 ? 1 : 2)) * 3 + (gw1 < 24 ? 0 : (gw1 < 36 ? 1 : 2));
        }
        for (int r = 0; r < 8; ++r) {
            size_t brow = bbase + (size_t)(r + 8 * half) * NTOK + kb * 32;
            float v0 = s0[r] + biasF[brow + lc];
            float v1 = s1[r] + biasF[brow + 16 + lc];
            if (shift) {
                if (qrid[r] != kr0) v0 -= 100.f;
                if (qrid[r] != kr1) v1 -= 100.f;
            }
            float pm = fmaxf(v0, v1);
            pm = fmaxf(pm, __shfl_xor(pm, 1, 32));
            pm = fmaxf(pm, __shfl_xor(pm, 2, 32));
            pm = fmaxf(pm, __shfl_xor(pm, 4, 32));
            pm = fmaxf(pm, __shfl_xor(pm, 8, 32));
            float mn = fmaxf(mx[r], pm);
            float co = __expf(mx[r] - mn);
            float e0 = __expf(v0 - mn), e1 = __expf(v1 - mn);
            float rs = e0 + e1;
            rs += __shfl_xor(rs, 1, 32); rs += __shfl_xor(rs, 2, 32);
            rs += __shfl_xor(rs, 4, 32); rs += __shfl_xor(rs, 8, 32);
            li[r] = li[r] * co + rs;
            mx[r] = mn;
            o0[r] *= co; o1[r] *= co;
            Pl[(r + 8 * half) * 32 + lc]      = f2bf(e0);   // C-layout -> row-major P in LDS
            Pl[(r + 8 * half) * 32 + 16 + lc] = f2bf(e1);
        }
        v16bf pA  = load_fragA(Pl + lc * 32 + half * 16);
        v16bf vB0 = load_fragB(vT + vbase + (size_t)lc        * NTOK + kb * 32 + half * 16);
        v16bf vB1 = load_fragB(vT + vbase + (size_t)(16 + lc) * NTOK + kb * 32 + half * 16);
        o0 = wmma_bf16(pA, vB0, o0);
        o1 = wmma_bf16(pA, vB1, o1);
    }
    for (int r = 0; r < 8; ++r) {
        float inv = 1.f / li[r];
        size_t orow = ((size_t)(w * NTOK + qt * 16 + r + 8 * half)) * DIM + h * HD;
        obuf[orow + lc]      = f2bf(o0[r] * inv);
        obuf[orow + 16 + lc] = f2bf(o1[r] * inv);
    }
}

// generic GEMM: C[M][Nout] = A[M][K](bf16) * Bw[Nout][K](bf16)^T + bias
// 32x64 tile per wave (2 A-frags x 4 B-frags -> 8 WMMA per K step)
// gelu=1 -> GELU + bf16 out, else f32 out
__global__ void k_gemm(const unsigned short* __restrict__ A, const unsigned short* __restrict__ Bw,
                       const float* __restrict__ bias, float* __restrict__ outF,
                       unsigned short* __restrict__ outB, int K, int Nout, int gelu) {
    int bn = blockIdx.x, bm = blockIdx.y;
    int lane = threadIdx.x, half = lane >> 4, lc = lane & 15;
    const unsigned short* arow0 = A + ((size_t)(bm * 32 + lc)) * K + half * 16;
    const unsigned short* arow1 = arow0 + (size_t)16 * K;
    const unsigned short* brow  = Bw + ((size_t)(bn * 64 + lc)) * K + half * 16;
    v8f acc[2][4];
    #pragma unroll
    for (int mi = 0; mi < 2; ++mi)
        #pragma unroll
        for (int j = 0; j < 4; ++j)
            #pragma unroll
            for (int r = 0; r < 8; ++r) acc[mi][j][r] = 0.f;
    for (int k0 = 0; k0 < K; k0 += 32) {
        v16bf a0 = load_fragA(arow0 + k0);
        v16bf a1 = load_fragA(arow1 + k0);
        #pragma unroll
        for (int j = 0; j < 4; ++j) {
            v16bf b = load_fragB(brow + (size_t)(j * 16) * K + k0);
            acc[0][j] = wmma_bf16(a0, b, acc[0][j]);
            acc[1][j] = wmma_bf16(a1, b, acc[1][j]);
        }
    }
    #pragma unroll
    for (int j = 0; j < 4; ++j) {
        float bj = bias[bn * 64 + j * 16 + lc];
        int col = bn * 64 + j * 16 + lc;
        #pragma unroll
        for (int mi = 0; mi < 2; ++mi)
            #pragma unroll
            for (int r = 0; r < 8; ++r) {
                size_t row = (size_t)(bm * 32 + mi * 16 + r + 8 * half);
                float v = acc[mi][j][r] + bj;
                if (gelu) {
                    v = 0.5f * v * (1.f + erff(v * 0.70710678f));
                    outB[row * Nout + col] = f2bf(v);
                } else {
                    outF[row * Nout + col] = v;
                }
            }
    }
}

// ---------------- layernorm / residual / head ----------------

// x += LN(src); mapped=1 -> src is in (window,token) order, apply window-reverse + roll(+s)
__global__ void k_ln_res(float* __restrict__ xbuf, const float* __restrict__ src,
                         const float* __restrict__ g, const float* __restrict__ bb,
                         int mapped, int s) {
    __shared__ float red[128];
    int t = blockIdx.x, tid = threadIdx.x;
    int srow;
    if (mapped) {
        int b = t / 2304, rr = t % 2304, gh = rr / WWID, gw = rr % WWID;
        int ph = (gh - s + HH) % HH, pw = (gw - s + WWID) % WWID;
        int win = b * 4 + (ph / WS) * 2 + (pw / WS);
        srow = win * NTOK + (ph % WS) * WS + (pw % WS);
    } else srow = t;
    const float* sp = src + (size_t)srow * DIM;
    float v0 = sp[tid], v1 = sp[tid + 128], v2 = sp[tid + 256];
    red[tid] = v0 + v1 + v2; __syncthreads();
    for (int off = 64; off > 0; off >>= 1) { if (tid < off) red[tid] += red[tid + off]; __syncthreads(); }
    float mu = red[0] / DIM; __syncthreads();
    float d0 = v0 - mu, d1 = v1 - mu, d2 = v2 - mu;
    red[tid] = d0 * d0 + d1 * d1 + d2 * d2; __syncthreads();
    for (int off = 64; off > 0; off >>= 1) { if (tid < off) red[tid] += red[tid + off]; __syncthreads(); }
    float rs = rsqrtf(red[0] / DIM + 1e-5f);
    float* xp = xbuf + (size_t)t * DIM;
    xp[tid]       += d0 * rs * g[tid]       + bb[tid];
    xp[tid + 128] += d1 * rs * g[tid + 128] + bb[tid + 128];
    xp[tid + 256] += d2 * rs * g[tid + 256] + bb[tid + 256];
}

// final LN + global average pool (atomic accumulate into feat)
__global__ void k_final_pool(const float* __restrict__ xbuf, const float* __restrict__ g,
                             const float* __restrict__ bb, float* __restrict__ feat) {
    __shared__ float red[128];
    int t = blockIdx.x, tid = threadIdx.x;
    const float* sp = xbuf + (size_t)t * DIM;
    float v0 = sp[tid], v1 = sp[tid + 128], v2 = sp[tid + 256];
    red[tid] = v0 + v1 + v2; __syncthreads();
    for (int off = 64; off > 0; off >>= 1) { if (tid < off) red[tid] += red[tid + off]; __syncthreads(); }
    float mu = red[0] / DIM; __syncthreads();
    float d0 = v0 - mu, d1 = v1 - mu, d2 = v2 - mu;
    red[tid] = d0 * d0 + d1 * d1 + d2 * d2; __syncthreads();
    for (int off = 64; off > 0; off >>= 1) { if (tid < off) red[tid] += red[tid + off]; __syncthreads(); }
    float rs = rsqrtf(red[0] / DIM + 1e-5f);
    int b = t / (HH * WWID);
    const float sc = 1.0f / (HH * WWID);
    atomicAdd(&feat[b * DIM + tid],       (d0 * rs * g[tid]       + bb[tid])       * sc);
    atomicAdd(&feat[b * DIM + tid + 128], (d1 * rs * g[tid + 128] + bb[tid + 128]) * sc);
    atomicAdd(&feat[b * DIM + tid + 256], (d2 * rs * g[tid + 256] + bb[tid + 256]) * sc);
}

__global__ void k_head(const float* __restrict__ feat, const float* __restrict__ hw,
                       const float* __restrict__ hb, float* __restrict__ out) {
    __shared__ float red[128];
    int b = blockIdx.x, tid = threadIdx.x;
    float s = feat[b * DIM + tid] * hw[tid] +
              feat[b * DIM + tid + 128] * hw[tid + 128] +
              feat[b * DIM + tid + 256] * hw[tid + 256];
    red[tid] = s; __syncthreads();
    for (int off = 64; off > 0; off >>= 1) { if (tid < off) red[tid] += red[tid + off]; __syncthreads(); }
    if (tid == 0) out[b] = red[0] + hb[0];
}

// ---------------- host orchestration ----------------

extern "C" void kernel_launch(void* const* d_in, const int* in_sizes, int n_in,
                              void* d_out, int out_size, void* d_ws, size_t ws_size,
                              hipStream_t stream) {
    (void)in_sizes; (void)n_in; (void)out_size; (void)ws_size;
    const float* x_in = (const float*)d_in[0];
    auto P = [&](int blk, int which) -> const float* {
        return (const float*)d_in[1 + blk * 16 + which];
    };
    const float* norm_g = (const float*)d_in[33];
    const float* norm_b = (const float*)d_in[34];
    const float* head_w = (const float*)d_in[35];
    const float* head_b = (const float*)d_in[36];

    char* wsb = (char*)d_ws; size_t off = 0;
    auto alloc = [&](size_t bytes) -> char* {
        char* p = wsb + off; off = (off + bytes + 255) & ~(size_t)255; return p;
    };

    unsigned short *wqkv[2], *wproj[2], *wfc1[2], *wfc2[2];
    for (int b = 0; b < 2; ++b) {
        wqkv[b] = (unsigned short*)alloc((size_t)1152 * 384 * 2);
        wproj[b] = (unsigned short*)alloc((size_t)384 * 384 * 2);
        wfc1[b] = (unsigned short*)alloc((size_t)1536 * 384 * 2);
        wfc2[b] = (unsigned short*)alloc((size_t)384 * 1536 * 2);
    }
    float* btab = (float*)alloc((size_t)TAB * HEADS * 4);
    float* xbuf = (float*)alloc((size_t)TOKS * DIM * 4);
    unsigned short* xbf = (unsigned short*)alloc((size_t)TOKS * DIM * 2);
    // qraw+kraw (attention stage) overlays hbuf (MLP stage) — identical size, disjoint lifetimes
    char* uni = alloc(2 * (size_t)TW * HEADS * NTOK * HD * 4);
    float* qraw = (float*)uni;
    float* kraw = (float*)(uni + (size_t)TW * HEADS * NTOK * HD * 4);
    unsigned short* hbuf = (unsigned short*)uni;
    unsigned short* vT = (unsigned short*)alloc((size_t)TW * HEADS * HD * NTOK * 2);
    unsigned short* qnb = (unsigned short*)alloc((size_t)TW * HEADS * NTOK * HD * 2);
    unsigned short* knb = (unsigned short*)alloc((size_t)TW * HEADS * NTOK * HD * 2);
    float* biasF = (float*)alloc((size_t)HEADS * NTOK * NTOK * 4);
    unsigned short* obuf = (unsigned short*)alloc((size_t)TW * NTOK * DIM * 2);
    float* tmp = (float*)alloc((size_t)TOKS * DIM * 4);
    float* feat = (float*)alloc((size_t)BATCH * DIM * 4);

    auto cvt = [&](const float* s, unsigned short* d, int n) {
        k_cvt_bf16<<<(n + 255) / 256, 256, 0, stream>>>(s, d, n);
    };
    for (int b = 0; b < 2; ++b) {
        cvt(P(b, 0),  wqkv[b], 1152 * 384);
        cvt(P(b, 6),  wproj[b], 384 * 384);
        cvt(P(b, 10), wfc1[b], 1536 * 384);
        cvt(P(b, 12), wfc2[b], 384 * 1536);
    }
    hipMemcpyAsync(xbuf, x_in, (size_t)TOKS * DIM * 4, hipMemcpyDeviceToDevice, stream);

    for (int blk = 0; blk < 2; ++blk) {
        int s = blk ? (WS / 2) : 0;
        k_cpb<<<(TAB + 63) / 64, 64, 0, stream>>>(P(blk, 3), P(blk, 4), P(blk, 5), btab);
        {
            int n = HEADS * NTOK * NTOK;
            k_bias_expand<<<(n + 255) / 256, 256, 0, stream>>>(btab, biasF);
        }
        {
            int n = TW * NTOK * DIM;
            k_stage_win<<<(n + 255) / 256, 256, 0, stream>>>(xbuf, xbf, s);
        }
        k_qkv<<<dim3(36, 288), 32, 0, stream>>>(xbf, wqkv[blk], P(blk, 1), qraw, kraw, vT);
        {
            int nrow = TW * HEADS * NTOK;
            k_norm_qk<<<(nrow + 127) / 128, 128, 0, stream>>>(qraw, qnb, P(blk, 2));
            k_norm_qk<<<(nrow + 127) / 128, 128, 0, stream>>>(kraw, knb, nullptr);
        }
        k_attn<<<dim3(36, HEADS, TW), 32, 0, stream>>>(qnb, knb, vT, biasF, obuf, s);
        k_gemm<<<dim3(DIM / 64, TOKS / 32), 32, 0, stream>>>(obuf, wproj[blk], P(blk, 7),
                                                             tmp, nullptr, DIM, DIM, 0);
        k_ln_res<<<TOKS, 128, 0, stream>>>(xbuf, tmp, P(blk, 8), P(blk, 9), 1, s);
        cvt(xbuf, xbf, TOKS * DIM);
        k_gemm<<<dim3(MLPD / 64, TOKS / 32), 32, 0, stream>>>(xbf, wfc1[blk], P(blk, 11),
                                                              nullptr, hbuf, DIM, MLPD, 1);
        k_gemm<<<dim3(DIM / 64, TOKS / 32), 32, 0, stream>>>(hbuf, wfc2[blk], P(blk, 13),
                                                             tmp, nullptr, MLPD, DIM, 0);
        k_ln_res<<<TOKS, 128, 0, stream>>>(xbuf, tmp, P(blk, 14), P(blk, 15), 0, 0);
    }
    hipMemsetAsync(feat, 0, (size_t)BATCH * DIM * 4, stream);
    k_final_pool<<<TOKS, 128, 0, stream>>>(xbuf, norm_g, norm_b, feat);
    k_head<<<BATCH, 128, 0, stream>>>(feat, head_w, head_b, (float*)d_out);
}
